// BasicBlock_88347477279404
// MI455X (gfx1250) — compile-verified, matrix operand order checked by reading, not run on hardware
//
#include <hip/hip_runtime.h>
#include <hip/hip_bf16.h>

// ---------------------------------------------------------------------------
// Fused ResNet BasicBlock for gfx1250 (MI455X):
//   conv3x3 -> BN -> per-elem act -> conv3x3 -> BN -> (+ per-elem act(x))
// Implicit GEMM via V_WMMA_F32_16X16X32_BF16, both convs fused in one kernel
// with an LDS-resident haloed intermediate tile (saves ~206 MB of h traffic).
// ---------------------------------------------------------------------------

typedef __attribute__((ext_vector_type(16))) __bf16 v16bf;
typedef __attribute__((ext_vector_type(8)))  float  v8f;

union FragU { v16bf v; uint4 q[2]; };

__device__ __forceinline__ v16bf load_frag16(const __bf16* p) {
  FragU u;
  u.q[0] = *(const uint4*)(p);
  u.q[1] = *(const uint4*)(p + 8);
  return u.v;
}

// 0=relu, 1=identity, 2=tanh, 3=sigmoid.
// Branch-free: 2x v_exp_f32 + 2x v_rcp_f32, numerically stable at the tails
// (exp overflow -> inf -> rcp -> 0 gives the correct saturation).
__device__ __forceinline__ float apply_act(float x, int code) {
  float relu = fmaxf(x, 0.0f);
  float ax = fabsf(x);
  float t  = __expf(-2.0f * ax);                       // in (0, 1]
  float th = __builtin_copysignf((1.0f - t) * __frcp_rn(1.0f + t), x);
  float sg = __frcp_rn(1.0f + __expf(-x));
  return (code == 0) ? relu : (code == 1) ? x : (code == 2) ? th : sg;
}

// ---------------------------------------------------------------------------
// Prep kernel: repack OIHW fp32 weights into WMMA A-fragment order (bf16)
// and precompute BN scale/shift vectors.
// Packed layout: wp[((mt*18 + t)*32 + lane)*16 + j]
//   mt   = 16-row block of output channels (4 blocks)
//   t    = k-step (tap-major: tap = t>>1, channel half = (t&1)*32)
//   lane = WMMA lane; lane<16 holds K(local)=0..15, lane>=16 holds 16..31
// ---------------------------------------------------------------------------
__global__ void prep_kernel(const float* __restrict__ w1, const float* __restrict__ w2,
                            const float* __restrict__ g1, const float* __restrict__ b1,
                            const float* __restrict__ m1, const float* __restrict__ v1,
                            const float* __restrict__ g2, const float* __restrict__ b2,
                            const float* __restrict__ m2, const float* __restrict__ v2,
                            __bf16* __restrict__ wp1, __bf16* __restrict__ wp2,
                            float* __restrict__ bn) {
  int tid = blockIdx.x * blockDim.x + threadIdx.x;
  if (tid < 2 * 36864) {
    int which = tid / 36864;
    int r     = tid - which * 36864;        // ((mt*18 + t)*32 + l)*16 + j
    int j  = r & 15;
    int l  = (r >> 4) & 31;
    int tt = (r >> 9) % 18;
    int mt = r / (18 * 512);
    int tap = tt >> 1;
    int c   = ((tt & 1) << 5) + ((l >> 4) << 4) + j;  // input channel
    int m   = (mt << 4) + (l & 15);                   // output channel
    const float* w = which ? w2 : w1;
    float val = w[(m * 64 + c) * 9 + tap];
    (which ? wp2 : wp1)[r] = (__bf16)val;
  }
  if (tid < 64) {
    float i1 = g1[tid] * rsqrtf(v1[tid] + 1e-5f);
    bn[tid]       = i1;
    bn[64 + tid]  = b1[tid] - m1[tid] * i1;
    float i2 = g2[tid] * rsqrtf(v2[tid] + 1e-5f);
    bn[128 + tid] = i2;
    bn[192 + tid] = b2[tid] - m2[tid] * i2;
  }
}

// ---------------------------------------------------------------------------
// Fused main kernel. Grid (7,14,32): 16x8 spatial output tile per workgroup.
// LDS: x tile 12x20 pos x 64ch (halo 2), h tile 10x18 pos x 64ch (halo 1).
// Channel-contiguous per position, stride 72 (=144B, 16B aligned, padded).
// ---------------------------------------------------------------------------
#define CS 72           // channel stride (elements) per LDS position
__global__ void __launch_bounds__(256)
fused_block_kernel(const float* __restrict__ x,
                   const __bf16* __restrict__ wp1, const __bf16* __restrict__ wp2,
                   const float* __restrict__ bn,
                   const int* __restrict__ codes_f, const int* __restrict__ codes_s,
                   float* __restrict__ out) {
  __shared__ __bf16 x_lds[240 * CS];   // 12x20 positions, 34,560 B
  __shared__ __bf16 h_lds[180 * CS];   // 10x18 positions, 25,920 B

  const int w0 = blockIdx.x * 16;
  const int h0 = blockIdx.y * 8;
  const int b  = blockIdx.z;
  const int lane = threadIdx.x & 31;
  const int wave = threadIdx.x >> 5;
  const int half = lane >> 4;
  const int khalf = half << 4;

  // ---- Phase 0: stage haloed x tile to LDS as bf16 (coalesced along w) ----
  for (int idx = threadIdx.x; idx < 64 * 240; idx += 256) {
    int c  = idx / 240;
    int p  = idx - c * 240;
    int ph = p / 20, pw = p - ph * 20;
    int gh = h0 - 2 + ph, gw = w0 - 2 + pw;
    float v = 0.0f;
    if ((unsigned)gh < 112u && (unsigned)gw < 112u)
      v = x[(((size_t)b * 64 + c) * 112 + gh) * 112 + gw];
    x_lds[p * CS + c] = (__bf16)v;
  }
  __syncthreads();

  // ---- Phase 1: conv1 GEMM -> BN1 -> per-elem act -> h tile (with halo) ---
  // D[m=outch 64, n=180 haloed positions]: 4 m-tiles x 12 n-tiles = 48 tiles.
  for (int tile = wave; tile < 48; tile += 8) {
    int mt = tile / 12, nt = tile - mt * 12;
    int n  = (nt << 4) + (lane & 15);
    int nc = n < 180 ? n : 179;                 // clamp tail lanes (discarded)
    int nh = nc / 18, nw = nc - nh * 18;        // position in 10x18 h region
    v8f acc = {0.f, 0.f, 0.f, 0.f, 0.f, 0.f, 0.f, 0.f};
#pragma unroll
    for (int t = 0; t < 18; ++t) {
      const int tap = t >> 1;
      const int rr = tap / 3, ss = tap - rr * 3;
      const int cb = ((t & 1) << 5) + khalf;
      v16bf a  = load_frag16(wp1 + ((((mt * 18 + t) << 5) + lane) << 4));
      v16bf bf = load_frag16(&x_lds[((nh + rr) * 20 + (nw + ss)) * CS + cb]);
      acc = __builtin_amdgcn_wmma_f32_16x16x32_bf16(
          false, a, false, bf, (short)0, acc, false, false);
    }
    int gh = h0 - 1 + nh, gw = w0 - 1 + nw;
    bool inimg = ((unsigned)gh < 112u) && ((unsigned)gw < 112u) && (n < 180);
#pragma unroll
    for (int v = 0; v < 8; ++v) {
      int cm = (mt << 4) + v + (half << 3);
      float val = 0.0f;
      if (inimg) {
        val = acc[v] * bn[cm] + bn[64 + cm];
        int code = codes_f[(cm * 112 + gh) * 112 + gw];
        val = apply_act(val, code);
      }
      if (n < 180) h_lds[n * CS + cm] = (__bf16)val;  // zero outside image
    }
  }
  __syncthreads();

  // ---- Phase 2: conv2 GEMM -> BN2 + activated shortcut + residual --------
  // D[m=outch 64, n=128 output positions]: 4 m-tiles x 8 n-tiles = 32 tiles.
  for (int tile = wave; tile < 32; tile += 8) {
    int mt = tile >> 3, nt = tile & 7;
    int n  = (nt << 4) + (lane & 15);
    int nh = n >> 4, nw = n & 15;               // position in 8x16 out tile
    v8f acc = {0.f, 0.f, 0.f, 0.f, 0.f, 0.f, 0.f, 0.f};
#pragma unroll
    for (int t = 0; t < 18; ++t) {
      const int tap = t >> 1;
      const int rr = tap / 3, ss = tap - rr * 3;
      const int cb = ((t & 1) << 5) + khalf;
      v16bf a  = load_frag16(wp2 + ((((mt * 18 + t) << 5) + lane) << 4));
      v16bf bf = load_frag16(&h_lds[((nh + rr) * 18 + (nw + ss)) * CS + cb]);
      acc = __builtin_amdgcn_wmma_f32_16x16x32_bf16(
          false, a, false, bf, (short)0, acc, false, false);
    }
    int gh = h0 + nh, gw = w0 + nw;
#pragma unroll
    for (int v = 0; v < 8; ++v) {
      int cm = (mt << 4) + v + (half << 3);
      float val = acc[v] * bn[128 + cm] + bn[192 + cm];
      float xv = (float)x_lds[((nh + 2) * 20 + (nw + 2)) * CS + cm];
      int code = codes_s[(cm * 112 + gh) * 112 + gw];
      val += apply_act(xv, code);
      out[(((size_t)b * 64 + cm) * 112 + gh) * 112 + gw] = val;  // coalesced
    }
  }
}

extern "C" void kernel_launch(void* const* d_in, const int* in_sizes, int n_in,
                              void* d_out, int out_size, void* d_ws, size_t ws_size,
                              hipStream_t stream) {
  const float* x  = (const float*)d_in[0];
  const float* w1 = (const float*)d_in[1];
  const float* w2 = (const float*)d_in[2];
  const float* g1 = (const float*)d_in[3];
  const float* b1 = (const float*)d_in[4];
  const float* m1 = (const float*)d_in[5];
  const float* v1 = (const float*)d_in[6];
  const float* g2 = (const float*)d_in[7];
  const float* b2 = (const float*)d_in[8];
  const float* m2 = (const float*)d_in[9];
  const float* v2 = (const float*)d_in[10];
  const int* cf = (const int*)d_in[11];
  const int* cs = (const int*)d_in[12];
  float* out = (float*)d_out;

  __bf16* wp1 = (__bf16*)d_ws;                       // 73,728 B
  __bf16* wp2 = wp1 + 36864;                         // 73,728 B
  float*  bn  = (float*)((char*)d_ws + 2 * 73728);   // 1,024 B

  prep_kernel<<<(2 * 36864 + 255) / 256, 256, 0, stream>>>(
      w1, w2, g1, b1, m1, v1, g2, b2, m2, v2, wp1, wp2, bn);

  fused_block_kernel<<<dim3(7, 14, 32), 256, 0, stream>>>(
      x, wp1, wp2, bn, cf, cs, out);
}
